// MHA_80659485819508
// MI455X (gfx1250) — matrix operand order repaired and checked by reading, not compiled, not run on hardware
//
#include <hip/hip_runtime.h>
#include <stddef.h>

// ---------------------------------------------------------------------------
// MI455X (gfx1250) fused 1x1conv + MHA
//   f16 WMMA (v_wmma_f32_16x16x32_f16) + Tensor Data Mover staging (TENSORcnt)
// ---------------------------------------------------------------------------
typedef _Float16 h16;
typedef __attribute__((ext_vector_type(16))) _Float16 v16h;
typedef __attribute__((ext_vector_type(8)))  _Float16 v8h;
typedef __attribute__((ext_vector_type(4)))  _Float16 v4h;
typedef __attribute__((ext_vector_type(8)))  float    v8f;
typedef __attribute__((ext_vector_type(4)))  unsigned int u32x4;
typedef __attribute__((ext_vector_type(8)))  int      i32x8;
typedef __attribute__((ext_vector_type(4)))  int      i32x4;

#define NB     8
#define CH     512
#define SP     1024
#define NQ     1024
#define NH     8
#define DK     512
#define QSCALE 0.044194173824159216f  // 1/sqrt(512)

// ---- WMMA fragment helpers (layouts per CDNA5 ISA 7.12.2) ------------------
__device__ __forceinline__ v16h frag_a(const h16* __restrict__ row, int hi) {
  v8h lo = *(const v8h*)(row + 8 * hi);
  v8h hp = *(const v8h*)(row + 16 + 8 * hi);
  return __builtin_shufflevector(lo, hp, 0,1,2,3,4,5,6,7,8,9,10,11,12,13,14,15);
}
__device__ __forceinline__ v16h frag_b(const h16* __restrict__ rowN, int hi) {
  v8h lo = *(const v8h*)(rowN + 8 * hi);
  v8h hp = *(const v8h*)(rowN + 16 + 8 * hi);
  return __builtin_shufflevector(lo, hp, 0,1,2,3,4,5,6,7,8,9,10,11,12,13,14,15);
}
__device__ __forceinline__ v8f wmma32(v16h a, v16h b, v8f c) {
  return __builtin_amdgcn_wmma_f32_16x16x32_f16(false, a, false, b, (short)0, c,
                                                false, false);
}

// ---- Tensor Data Mover: 2-D f16 tile -> LDS, padded to pitch 40 f16 --------
// D# per ISA ch.8: group0 {count=1, lds_addr, global_addr, type=2},
// group1 {data_size=2B, pad 4 DW per 16 DW, tensor==tile dims, stride0}.
__device__ __forceinline__ void tdm_load_f16_tile(
    const h16* gptr, unsigned int lds_byte_off,
    unsigned int tile_d0 /*cols, f16*/, unsigned int tile_d1 /*rows*/,
    unsigned long long row_stride_elems) {
  unsigned long long ga = (unsigned long long)gptr;
  u32x4 g0;
  g0[0] = 1u;                                   // count=1 (valid), user mode
  g0[1] = lds_byte_off;                         // lds_addr
  g0[2] = (unsigned int)(ga & 0xFFFFFFFFu);     // global_addr[31:0]
  g0[3] = (unsigned int)((ga >> 32) & 0x01FFFFFFu) | (2u << 30);  // type=2
  i32x8 g1;
  g1[0] = (int)((1u << 16)      // data_size = 2 bytes
              | (1u << 20)      // pad_enable
              | (3u << 22)      // pad_interval: 16 dwords (one 32-f16 row)
              | (3u << 25));    // pad_amount: 4 dwords -> pitch 40 f16
  g1[1] = (int)((tile_d0 & 0xFFFFu) << 16);               // tensor_dim0 lo
  g1[2] = (int)((tile_d0 >> 16) | ((tile_d1 & 0xFFFFu) << 16));  // td0 hi|td1 lo
  g1[3] = (int)((tile_d1 >> 16) | (tile_d0 << 16));       // td1 hi | tile_dim0
  g1[4] = (int)(tile_d1);                                 // tile_dim1 (dim2=0)
  g1[5] = (int)(row_stride_elems & 0xFFFFFFFFu);          // stride0 lo
  g1[6] = (int)((row_stride_elems >> 32) & 0xFFFFu);      // stride0 hi|stride1=0
  g1[7] = 0;
  i32x4 z4 = {0, 0, 0, 0};
#if __clang_major__ >= 23
  i32x8 z8 = {0, 0, 0, 0, 0, 0, 0, 0};
  __builtin_amdgcn_tensor_load_to_lds(g0, g1, z4, z4, z8, 0);
#else
  __builtin_amdgcn_tensor_load_to_lds(g0, g1, z4, z4, 0);
#endif
}
__device__ __forceinline__ unsigned int lds_off_of(const void* p) {
  return (unsigned int)(unsigned long long)p;  // LDS aperture: low 32 = offset
}

// ---------------------------------------------------------------------------
// K0: f32 -> f16 weight conversion (one-shot, makes weights TDM-able)
// ---------------------------------------------------------------------------
__global__ __launch_bounds__(256) void k_cvt_f16(const float* __restrict__ src,
                                                 h16* __restrict__ dst, int n) {
  int i = (blockIdx.x * 256 + threadIdx.x) * 4;
  if (i < n) {
    float4 v = *(const float4*)(src + i);
    v4h o = {(h16)v.x, (h16)v.y, (h16)v.z, (h16)v.w};
    *(v4h*)(dst + i) = o;
  }
}

// ---------------------------------------------------------------------------
// K1: tok from 1x1 conv.  T = conv_w(512x512) @ X_b(512x1024), .view remap.
// f32 sources (needs convert+transpose) -> classic LDS staging.
// ---------------------------------------------------------------------------
__global__ __launch_bounds__(256) void k_conv_tok(
    const float* __restrict__ x, const float* __restrict__ conv_w,
    const float* __restrict__ conv_b, h16* __restrict__ tok) {
  const int b = blockIdx.z;
  const int m0 = blockIdx.x * 64;
  const int n0 = blockIdx.y * 128;
  __shared__ h16 As[64][40];
  __shared__ h16 Bs[128][48];
  const int tid = threadIdx.x, lane = tid & 31, wid = tid >> 5;
  const int wr = wid >> 2, wc = wid & 3;
  const int mlo = lane & 15, hi = lane >> 4;

  v8f acc[2][2] = {};
  for (int kk = 0; kk < CH; kk += 32) {
    __syncthreads();
    {
      int r = tid >> 2, c = (tid & 3) * 8;
      const float* g = conv_w + (size_t)(m0 + r) * CH + kk + c;
#pragma unroll
      for (int i = 0; i < 8; ++i) As[r][c + i] = (h16)g[i];
    }
    {
      int c = tid >> 3, sb = (tid & 7) * 16;
      const float* g = x + ((size_t)b * CH + kk + c) * SP + n0 + sb;
#pragma unroll
      for (int i = 0; i < 16; ++i) Bs[sb + i][c] = (h16)g[i];
    }
    __syncthreads();
#pragma unroll
    for (int mi = 0; mi < 2; ++mi) {
      v16h a = frag_a(&As[wr * 32 + mi * 16 + mlo][0], hi);
#pragma unroll
      for (int nj = 0; nj < 2; ++nj) {
        v16h bf = frag_b(&Bs[wc * 32 + nj * 16 + mlo][0], hi);
        acc[mi][nj] = wmma32(a, bf, acc[mi][nj]);
      }
    }
  }
#pragma unroll
  for (int mi = 0; mi < 2; ++mi)
#pragma unroll
    for (int nj = 0; nj < 2; ++nj)
#pragma unroll
      for (int g = 0; g < 8; ++g) {
        int o = m0 + wr * 32 + mi * 16 + g + 8 * hi;
        int s = n0 + wc * 32 + nj * 16 + mlo;
        float v = acc[mi][nj][g] + conv_b[o];
        int n = o * 2 + (s >> 9);
        tok[((size_t)b * NQ + n) * CH + (s & 511)] = (h16)v;
      }
}

// ---------------------------------------------------------------------------
// K2: Q/K/V projections, TDM double-buffered staging.
// Y = tok_b(1024x512) @ W16^T + bias.  mode 0:Q(scaled) 1:K 2:V(transposed)
// ---------------------------------------------------------------------------
__global__ __launch_bounds__(256) void k_proj_qkv(
    const h16* __restrict__ tok, const h16* __restrict__ W16,
    const float* __restrict__ bias, h16* __restrict__ dst, int mode) {
  const int b = blockIdx.z;
  const int m0 = blockIdx.x * 64;
  const int n0 = blockIdx.y * 128;
  __shared__ h16 As[2][64][40];
  __shared__ h16 Bs[2][128][40];
  const int tid = threadIdx.x, lane = tid & 31, wid = tid >> 5;
  const int wr = wid >> 2, wc = wid & 3;
  const int mlo = lane & 15, hi = lane >> 4;
  const h16* Ag = tok + ((size_t)b * NQ + m0) * CH;
  const h16* Bg = W16 + (size_t)n0 * CH;

  if (tid == 0) {  // prologue: stage step 0 (TDM ignores EXEC; one wave issues)
    tdm_load_f16_tile(Ag, lds_off_of(&As[0][0][0]), 32, 64, CH);
    tdm_load_f16_tile(Bg, lds_off_of(&Bs[0][0][0]), 32, 128, CH);
  }
  v8f acc[2][2] = {};
  const int nsteps = CH / 32;
  for (int s = 0; s < nsteps; ++s) {
    const int cur = s & 1;
    if (tid == 0) {
      if (s + 1 < nsteps) {
        tdm_load_f16_tile(Ag + (s + 1) * 32, lds_off_of(&As[cur ^ 1][0][0]),
                          32, 64, CH);
        tdm_load_f16_tile(Bg + (s + 1) * 32, lds_off_of(&Bs[cur ^ 1][0][0]),
                          32, 128, CH);
        __builtin_amdgcn_s_wait_tensorcnt(2);  // retire older pair only
      } else {
        __builtin_amdgcn_s_wait_tensorcnt(0);
      }
    }
    __syncthreads();
#pragma unroll
    for (int mi = 0; mi < 2; ++mi) {
      v16h a = frag_a(&As[cur][wr * 32 + mi * 16 + mlo][0], hi);
#pragma unroll
      for (int nj = 0; nj < 2; ++nj) {
        v16h bf = frag_b(&Bs[cur][wc * 32 + nj * 16 + mlo][0], hi);
        acc[mi][nj] = wmma32(a, bf, acc[mi][nj]);
      }
    }
    __syncthreads();
  }
#pragma unroll
  for (int mi = 0; mi < 2; ++mi)
#pragma unroll
    for (int nj = 0; nj < 2; ++nj)
#pragma unroll
      for (int g = 0; g < 8; ++g) {
        int m = m0 + wr * 32 + mi * 16 + g + 8 * hi;
        int o = n0 + wc * 32 + nj * 16 + mlo;
        float v = acc[mi][nj][g] + bias[o];
        int h = o >> 9, dd = o & 511;
        if (mode == 0) {
          v *= QSCALE;
          dst[(((size_t)b * NH + h) * NQ + m) * DK + dd] = (h16)v;
        } else if (mode == 1) {
          dst[(((size_t)b * NH + h) * NQ + m) * DK + dd] = (h16)v;
        } else {
          dst[(((size_t)b * NH + h) * DK + dd) * NQ + m] = (h16)v;  // V^T
        }
      }
}

// ---------------------------------------------------------------------------
// K3: flash attention, 4 waves / (b,h,q-tile of 32), nk chunk 32.
// S from L2-resident Q/K global fragments while the TDM streams the V^T chunk
// (512x32 f16) into LDS; online softmax; PV from LDS.
// ---------------------------------------------------------------------------
__global__ __launch_bounds__(128) void k_flash(
    const h16* __restrict__ Q, const h16* __restrict__ Km,
    const h16* __restrict__ VT, h16* __restrict__ O) {
  const int bh = blockIdx.y;
  const int b = bh >> 3, h = bh & 7;
  const int q0 = blockIdx.x * 32;
  const int tid = threadIdx.x, lane = tid & 31, w = tid >> 5;
  const int mlo = lane & 15, hi = lane >> 4;
  const h16* Qp = Q + ((size_t)bh * NQ + q0) * DK;
  const h16* Kp = Km + (size_t)bh * NQ * DK;
  const h16* Vh = VT + (size_t)bh * DK * NQ;  // [512 dv][1024 nk]
  __shared__ h16 Vlds[512][40];               // V^T chunk, TDM-padded pitch
  __shared__ h16 Plds[32][40];

  v8f oacc[2][8] = {};
  float rowm[2][8], rowl[2][8];
#pragma unroll
  for (int mi = 0; mi < 2; ++mi)
#pragma unroll
    for (int g = 0; g < 8; ++g) { rowm[mi][g] = -1e30f; rowl[mi][g] = 0.f; }

  for (int nk0 = 0; nk0 < NQ; nk0 += 32) {
    if (tid == 0)  // async V^T chunk DMA, overlapped with the S phase below
      tdm_load_f16_tile(Vh + nk0, lds_off_of(&Vlds[0][0]), 32, 512, NQ);
    // ---- S = Q K^T (pre-scaled) ----
    v8f S[2][2] = {};
    for (int kk = 0; kk < DK; kk += 32) {
      v16h a[2];
#pragma unroll
      for (int mi = 0; mi < 2; ++mi)
        a[mi] = frag_a(Qp + (size_t)(mi * 16 + mlo) * DK + kk, hi);
#pragma unroll
      for (int nj = 0; nj < 2; ++nj) {
        v16h bf = frag_b(Kp + (size_t)(nk0 + nj * 16 + mlo) * DK + kk, hi);
#pragma unroll
        for (int mi = 0; mi < 2; ++mi) S[mi][nj] = wmma32(a[mi], bf, S[mi][nj]);
      }
    }
    // ---- online softmax (row = mi*16 + g + 8*hi) ----
#pragma unroll
    for (int mi = 0; mi < 2; ++mi) {
#pragma unroll
      for (int g = 0; g < 8; ++g) {
        float mx = fmaxf(S[mi][0][g], S[mi][1][g]);
        mx = fmaxf(mx, __shfl_xor(mx, 1, 32));
        mx = fmaxf(mx, __shfl_xor(mx, 2, 32));
        mx = fmaxf(mx, __shfl_xor(mx, 4, 32));
        mx = fmaxf(mx, __shfl_xor(mx, 8, 32));
        float mnew = fmaxf(rowm[mi][g], mx);
        float alpha = __expf(rowm[mi][g] - mnew);
        rowm[mi][g] = mnew;
        float rs = 0.f;
#pragma unroll
        for (int nj = 0; nj < 2; ++nj) {
          float p = __expf(S[mi][nj][g] - mnew);
          S[mi][nj][g] = p;
          rs += p;
        }
        rs += __shfl_xor(rs, 1, 32);
        rs += __shfl_xor(rs, 2, 32);
        rs += __shfl_xor(rs, 4, 32);
        rs += __shfl_xor(rs, 8, 32);
        rowl[mi][g] = rowl[mi][g] * alpha + rs;
#pragma unroll
        for (int jn = 0; jn < 8; ++jn) oacc[mi][jn][g] *= alpha;
      }
    }
    if (tid == 0) __builtin_amdgcn_s_wait_tensorcnt(0);  // V^T chunk landed
    if (w < 2) {  // P (C-layout) -> LDS (A-layout source); wave w owns cols
#pragma unroll
      for (int mi = 0; mi < 2; ++mi)
#pragma unroll
        for (int g = 0; g < 8; ++g)
          Plds[mi * 16 + g + 8 * hi][w * 16 + mlo] = (h16)S[mi][w][g];
    }
    __syncthreads();
    // ---- O += P @ V from LDS (wave w owns dv slice w*128..) ----
    {
      v16h a[2];
#pragma unroll
      for (int mi = 0; mi < 2; ++mi) a[mi] = frag_a(&Plds[mi * 16 + mlo][0], hi);
#pragma unroll
      for (int jn = 0; jn < 8; ++jn) {
        v16h bf = frag_b(&Vlds[w * 128 + jn * 16 + mlo][0], hi);
#pragma unroll
        for (int mi = 0; mi < 2; ++mi)
          oacc[mi][jn] = wmma32(a[mi], bf, oacc[mi][jn]);
      }
    }
    __syncthreads();
  }
  h16* Op = O + ((size_t)b * NQ + q0) * (NH * DK) + h * DK + w * 128;
#pragma unroll
  for (int mi = 0; mi < 2; ++mi)
#pragma unroll
    for (int jn = 0; jn < 8; ++jn)
#pragma unroll
      for (int g = 0; g < 8; ++g) {
        float v = oacc[mi][jn][g] / rowl[mi][g];
        Op[(size_t)(mi * 16 + g + 8 * hi) * (NH * DK) + jn * 16 + mlo] = (h16)v;
      }
}

// ---------------------------------------------------------------------------
// K4: out = O(8192x4096) @ wo16^T + bo, TDM double-buffered, f32 .view store.
// ---------------------------------------------------------------------------
__global__ __launch_bounds__(256) void k_out_proj(
    const h16* __restrict__ Oc, const h16* __restrict__ wo16,
    const float* __restrict__ bo, float* __restrict__ out) {
  const int m0 = blockIdx.x * 64;
  const int n0 = blockIdx.y * 128;
  __shared__ h16 As[2][64][40];
  __shared__ h16 Bs[2][128][40];
  const int tid = threadIdx.x, lane = tid & 31, wid = tid >> 5;
  const int wr = wid >> 2, wc = wid & 3;
  const int mlo = lane & 15, hi = lane >> 4;
  const int KT = NH * DK;  // 4096
  const h16* Ag = Oc + (size_t)m0 * KT;
  const h16* Bg = wo16 + (size_t)n0 * KT;

  if (tid == 0) {
    tdm_load_f16_tile(Ag, lds_off_of(&As[0][0][0]), 32, 64, KT);
    tdm_load_f16_tile(Bg, lds_off_of(&Bs[0][0][0]), 32, 128, KT);
  }
  v8f acc[2][2] = {};
  const int nsteps = KT / 32;
  for (int s = 0; s < nsteps; ++s) {
    const int cur = s & 1;
    if (tid == 0) {
      if (s + 1 < nsteps) {
        tdm_load_f16_tile(Ag + (s + 1) * 32, lds_off_of(&As[cur ^ 1][0][0]),
                          32, 64, KT);
        tdm_load_f16_tile(Bg + (s + 1) * 32, lds_off_of(&Bs[cur ^ 1][0][0]),
                          32, 128, KT);
        __builtin_amdgcn_s_wait_tensorcnt(2);
      } else {
        __builtin_amdgcn_s_wait_tensorcnt(0);
      }
    }
    __syncthreads();
#pragma unroll
    for (int mi = 0; mi < 2; ++mi) {
      v16h a = frag_a(&As[cur][wr * 32 + mi * 16 + mlo][0], hi);
#pragma unroll
      for (int nj = 0; nj < 2; ++nj) {
        v16h bf = frag_b(&Bs[cur][wc * 32 + nj * 16 + mlo][0], hi);
        acc[mi][nj] = wmma32(a, bf, acc[mi][nj]);
      }
    }
    __syncthreads();
  }
#pragma unroll
  for (int mi = 0; mi < 2; ++mi)
#pragma unroll
    for (int nj = 0; nj < 2; ++nj)
#pragma unroll
      for (int g = 0; g < 8; ++g) {
        int m = m0 + wr * 32 + mi * 16 + g + 8 * hi;
        int o = n0 + wc * 32 + nj * 16 + mlo;
        float v = acc[mi][nj][g] + bo[o];
        int b = m >> 10, nq = m & 1023;
        int c = nq >> 1, sidx = ((nq & 1) << 9) + o;
        out[((size_t)b * CH + c) * SP + sidx] = v;
      }
}

// ---------------------------------------------------------------------------
extern "C" void kernel_launch(void* const* d_in, const int* in_sizes, int n_in,
                              void* d_out, int out_size, void* d_ws,
                              size_t ws_size, hipStream_t stream) {
  const float* x      = (const float*)d_in[0];
  const float* conv_w = (const float*)d_in[1];
  const float* conv_b = (const float*)d_in[2];
  const float* wq = (const float*)d_in[3];
  const float* bq = (const float*)d_in[4];
  const float* wk = (const float*)d_in[5];
  const float* bk = (const float*)d_in[6];
  const float* wv = (const float*)d_in[7];
  const float* bv = (const float*)d_in[8];
  const float* wo = (const float*)d_in[9];
  const float* bo = (const float*)d_in[10];
  float* out = (float*)d_out;

  // ws (f16): tok 8M | Q 64M | K 64M | V^T 64M | O 64M | Wq/Wk/Wv/Wo16 4M each
  char* ws = (char*)d_ws;
  h16* tok  = (h16*)(ws);
  h16* Qb   = (h16*)(ws + (8ull << 20));
  h16* Kb   = (h16*)(ws + (72ull << 20));
  h16* VTb  = (h16*)(ws + (136ull << 20));
  h16* Ob   = (h16*)(ws + (200ull << 20));
  h16* Wq16 = (h16*)(ws + (264ull << 20));
  h16* Wk16 = (h16*)(ws + (268ull << 20));
  h16* Wv16 = (h16*)(ws + (272ull << 20));
  h16* Wo16 = (h16*)(ws + (276ull << 20));

  const int NW = NH * DK * CH;  // 2M elements per projection weight
  dim3 blk(256);
  k_cvt_f16<<<dim3(NW / 1024), blk, 0, stream>>>(wq, Wq16, NW);
  k_cvt_f16<<<dim3(NW / 1024), blk, 0, stream>>>(wk, Wk16, NW);
  k_cvt_f16<<<dim3(NW / 1024), blk, 0, stream>>>(wv, Wv16, NW);
  k_cvt_f16<<<dim3(NW / 1024), blk, 0, stream>>>(wo, Wo16, NW);
  k_conv_tok<<<dim3(8, 8, NB), blk, 0, stream>>>(x, conv_w, conv_b, tok);
  k_proj_qkv<<<dim3(16, 32, NB), blk, 0, stream>>>(tok, Wq16, bq, Qb, 0);
  k_proj_qkv<<<dim3(16, 32, NB), blk, 0, stream>>>(tok, Wk16, bk, Kb, 1);
  k_proj_qkv<<<dim3(16, 32, NB), blk, 0, stream>>>(tok, Wv16, bv, VTb, 2);
  k_flash<<<dim3(NQ / 32, NB * NH), dim3(128), 0, stream>>>(Qb, Kb, VTb, Ob);
  k_out_proj<<<dim3(128, 4), blk, 0, stream>>>(Ob, Wo16, bo, out);
}
